// FML_30133490549669
// MI455X (gfx1250) — matrix operand order repaired and checked by reading, not compiled, not run on hardware
//
#include <hip/hip_runtime.h>
#include <hip/hip_bf16.h>

#define SCALE_S 64.0f
#define K_STEP 32
#define M_BLK 64
#define N_BLK 256

typedef __attribute__((ext_vector_type(16))) __bf16 v16bf;
typedef __attribute__((ext_vector_type(8)))  float  v8f;

__device__ __forceinline__ unsigned short f2bf(float f) {
    unsigned u = __float_as_uint(f);
    unsigned r = (u + 0x7fffu + ((u >> 16) & 1u)) >> 16;
    return (unsigned short)r;
}
__device__ __forceinline__ float bf2f(unsigned short h) {
    return __uint_as_float(((unsigned)h) << 16);
}

// ---- CDNA5 async global->LDS copies (ASYNCcnt-tracked, no VGPR round trip) ----
// INST_OFFSET applies to BOTH the global and LDS addresses (ISA §10/15.18), so
// segments share one base register pair and differ only by the immediate.
template <int OFF>
__device__ __forceinline__ void async_copy_b128_off(const unsigned short* gsrc, unsigned lds) {
    asm volatile("global_load_async_to_lds_b128 %0, %1, off offset:%2"
                 :: "v"(lds), "v"(gsrc), "n"(OFF) : "memory");
}
__device__ __forceinline__ void wait_async0() {
    asm volatile("s_wait_asynccnt 0x0" ::: "memory");
}

// ---------------- Kernel 1: per-row L2 normalize, f32 -> bf16 ----------------
__global__ __launch_bounds__(128)
void k_normalize_rows(const float* __restrict__ src, unsigned short* __restrict__ dst, int D) {
    const int row = blockIdx.x;
    const int tid = threadIdx.x;
    __shared__ float red[4];
    const float* p = src + (size_t)row * D;
    float ss = 0.f;
    for (int k = tid; k < D; k += 128) { float v = p[k]; ss += v * v; }
    for (int off = 16; off; off >>= 1) ss += __shfl_xor(ss, off, 32);
    if ((tid & 31) == 0) red[tid >> 5] = ss;
    __syncthreads();
    ss = red[0] + red[1] + red[2] + red[3];
    float inv = rsqrtf(fmaxf(ss, 1e-12f));
    unsigned short* q = dst + (size_t)row * D;
    for (int k = tid; k < D; k += 128) q[k] = f2bf(p[k] * inv);
}

// ---------------- Kernel 2: zero row-sum accumulator ----------------
__global__ void k_zero(float* __restrict__ p, int n) {
    int i = blockIdx.x * blockDim.x + threadIdx.x;
    if (i < n) p[i] = 0.f;
}

// ---------------- Kernel 3: bf16 WMMA GEMM, async double-buffered LDS ----------------
// Block: 256 threads = 8 waves, block tile 64(M) x 256(N).
// Wave tile: 32x64 = 2 A-frags x 4 B-frags = 8 WMMA accumulators per K-step.
__global__ __launch_bounds__(256)
void k_cosface_gemm(const unsigned short* __restrict__ xb,   // [B,D] bf16 bits (normalized)
                    const unsigned short* __restrict__ wb,   // [C,D] bf16 bits (normalized)
                    float* __restrict__ out,                 // [B,C]
                    float* __restrict__ rowsum,              // [B]
                    int B, int C, int D) {
    __shared__ unsigned short lA[2][M_BLK][K_STEP];   // 2 x 4 KB
    __shared__ unsigned short lB[2][N_BLK][K_STEP];   // 2 x 16 KB, stored [class][k]

    const int tid  = threadIdx.x;
    const int lane = tid & 31;
    const int wave = tid >> 5;
    const int wm   = wave >> 2;          // 0..1  (M sub-tile, 32 rows)
    const int wn   = wave & 3;           // 0..3  (N sub-tile, 64 wide)
    const int l15  = lane & 15;
    const int half = lane >> 4;

    const int b0 = blockIdx.y * M_BLK;
    const int c0 = blockIdx.x * N_BLK;

    // Per-thread staging assignments (constant across K steps)
    const int ar = tid >> 2;             // A: 0..63 rows
    const int as = tid & 3;              // A: 4 segments of 8 bf16 (16 B)
    int arow = b0 + ar; if (arow >= B) arow = B - 1;
    const unsigned short* aSrc = xb + (size_t)arow * D + as * 8;

    const int br = tid;                  // B: 0..255 rows, 64 bytes each
    int bcol = c0 + br; if (bcol >= C) bcol = C - 1;
    const unsigned short* bSrc = wb + (size_t)bcol * D;

    // Loop-invariant LDS destination bases (generic LDS addr[31:0] == LDS offset)
    const unsigned ldsA[2] = { (unsigned)(uintptr_t)&lA[0][ar][as * 8],
                               (unsigned)(uintptr_t)&lA[1][ar][as * 8] };
    const unsigned ldsB[2] = { (unsigned)(uintptr_t)&lB[0][br][0],
                               (unsigned)(uintptr_t)&lB[1][br][0] };

    v8f acc[2][4] = {{{}, {}, {}, {}}, {{}, {}, {}, {}}};

    const int nsteps = D / K_STEP;

    // Prologue: fill buffer 0
    async_copy_b128_off<0 >(aSrc, ldsA[0]);
    async_copy_b128_off<0 >(bSrc, ldsB[0]);
    async_copy_b128_off<16>(bSrc, ldsB[0]);
    async_copy_b128_off<32>(bSrc, ldsB[0]);
    async_copy_b128_off<48>(bSrc, ldsB[0]);
    wait_async0();
    __syncthreads();

    for (int i = 0; i < nsteps; ++i) {
        const int cur = i & 1;
        const int nxt = cur ^ 1;
        // Kick off next stage's async fills (lands while we compute on `cur`)
        if (i + 1 < nsteps) {
            const int k1 = (i + 1) * K_STEP;
            async_copy_b128_off<0 >(aSrc + k1, ldsA[nxt]);
            const unsigned short* bs = bSrc + k1;
            async_copy_b128_off<0 >(bs, ldsB[nxt]);
            async_copy_b128_off<16>(bs, ldsB[nxt]);
            async_copy_b128_off<32>(bs, ldsB[nxt]);
            async_copy_b128_off<48>(bs, ldsB[nxt]);
        }

        // A fragments (16x32 bf16): lane holds M = wm*32 + a*16 + l15,
        // contiguous k runs [8*half, +8) and [16+8*half, +8)
        union Fr { v16bf v; uint4 q[2]; };
        Fr fa[2], fb[4];
        #pragma unroll
        for (int a = 0; a < 2; ++a) {
            int m = wm * 32 + a * 16 + l15;
            fa[a].q[0] = *(const uint4*)&lA[cur][m][8 * half];
            fa[a].q[1] = *(const uint4*)&lA[cur][m][16 + 8 * half];
        }
        // B fragments (32x16 bf16): lane holds N = l15, k runs [8*half, +8) and [16+8*half, +8)
        #pragma unroll
        for (int t = 0; t < 4; ++t) {
            int n = wn * 64 + t * 16 + l15;
            fb[t].q[0] = *(const uint4*)&lB[cur][n][8 * half];
            fb[t].q[1] = *(const uint4*)&lB[cur][n][16 + 8 * half];
        }
        #pragma unroll
        for (int a = 0; a < 2; ++a)
            #pragma unroll
            for (int t = 0; t < 4; ++t)
                acc[a][t] = __builtin_amdgcn_wmma_f32_16x16x32_bf16(
                    false, fa[a].v, false, fb[t].v, (short)0, acc[a][t], false, false);

        wait_async0();       // my fills for `nxt` landed
        __syncthreads();     // everyone done reading `cur` and filling `nxt`
    }

    // Epilogue: D layout — VGPR r: row = r + 8*half; lane l15 = col.
    #pragma unroll
    for (int a = 0; a < 2; ++a) {
        #pragma unroll
        for (int t = 0; t < 4; ++t) {
            int col = c0 + wn * 64 + t * 16 + l15;
            bool cvalid = (col < C);
            #pragma unroll
            for (int r = 0; r < 8; ++r) {
                int row = b0 + wm * 32 + a * 16 + half * 8 + r;
                bool valid = cvalid && (row < B);
                float cs = acc[a][t][r];
                if (valid)  // streaming store: never re-read, keep it out of L2
                    __builtin_nontemporal_store(cs * SCALE_S, &out[(size_t)row * C + col]);
                // per-row sum of exp(cosine): reduce the 16 lanes of this half-wave
                float e = valid ? __expf(cs) : 0.f;
                e += __shfl_xor(e, 8, 16);
                e += __shfl_xor(e, 4, 16);
                e += __shfl_xor(e, 2, 16);
                e += __shfl_xor(e, 1, 16);
                if (l15 == 0 && row < B) atomicAdd(&rowsum[row], e);
            }
        }
    }
}

// ---------------- Kernel 4: fix up the label column of each row ----------------
__global__ __launch_bounds__(32)
void k_fixup(const unsigned short* __restrict__ xb, const unsigned short* __restrict__ wb,
             const int* __restrict__ label, const float* __restrict__ rowsum,
             float* __restrict__ out, int C, int D) {
    const int b = blockIdx.x;
    const int lane = threadIdx.x;
    const int lbl = label[b];
    float s = 0.f;
    const unsigned short* xp = xb + (size_t)b * D;
    const unsigned short* wp = wb + (size_t)lbl * D;
    for (int k = lane; k < D; k += 32) s += bf2f(xp[k]) * bf2f(wp[k]);
    for (int off = 16; off; off >>= 1) s += __shfl_xor(s, off, 32);
    if (lane == 0) {
        float p = __expf(s) / rowsum[b];          // softmax prob at true class
        float margin = 0.5f * __expf(-p);
        out[(size_t)b * C + lbl] = (s - margin) * SCALE_S;
    }
}

extern "C" void kernel_launch(void* const* d_in, const int* in_sizes, int n_in,
                              void* d_out, int out_size, void* d_ws, size_t ws_size,
                              hipStream_t stream) {
    const float* x = (const float*)d_in[0];
    const float* w = (const float*)d_in[1];
    const int*   label = (const int*)d_in[2];
    float* out = (float*)d_out;

    const int B = in_sizes[2];            // 2048
    const int D = in_sizes[0] / B;        // 512
    const int C = in_sizes[1] / D;        // 100000

    unsigned short* xb = (unsigned short*)d_ws;          // [B,D] bf16
    unsigned short* wb = xb + (size_t)B * D;             // [C,D] bf16
    float* rowsum = (float*)(wb + (size_t)C * D);        // [B] f32

    k_normalize_rows<<<B, 128, 0, stream>>>(x, xb, D);
    k_normalize_rows<<<C, 128, 0, stream>>>(w, wb, D);
    k_zero<<<(B + 255) / 256, 256, 0, stream>>>(rowsum, B);

    dim3 grid((C + N_BLK - 1) / N_BLK, (B + M_BLK - 1) / M_BLK);
    k_cosface_gemm<<<grid, 256, 0, stream>>>(xb, wb, out, rowsum, B, C, D);

    k_fixup<<<B, 32, 0, stream>>>(xb, wb, label, rowsum, out, C, D);
}
